// Attention_18932215841016
// MI455X (gfx1250) — matrix operand order repaired and checked by reading, not compiled, not run on hardware
//
#include <hip/hip_runtime.h>
#include <hip/hip_bf16.h>
#include <math.h>

// ---------------------------------------------------------------------------
// MI455X (gfx1250) fused Bahdanau attention.
// k_proj GEMM (65536 x 1024 x 512 = 68.7 GFLOP) done with bf16 WMMA so the
// kernel is HBM/L2-bound. Each WG handles one batch x 64 s-rows so the W2
// matrix (resident in 192MB L2) is re-read only 1024x (1GB L2 traffic)
// instead of 4096x, and B fragments are register-reused across 4 s-tiles.
// tanh/Vw-dot epilogue fused; 128MB k_proj never materialized.
// ---------------------------------------------------------------------------

typedef __attribute__((ext_vector_type(16))) __bf16 v16bf;
typedef __attribute__((ext_vector_type(8)))  __bf16 v8bf;
typedef __attribute__((ext_vector_type(8)))  float  v8f;

#define NB   256      // batch
#define NS   256      // enc seq len
#define NU   512      // units
#define KE   1024     // 2*HIDDEN (enc feature dim)
#define KQ   2048     // 4*HIDDEN (query dim)
#define AROW 1032     // LDS row stride in bf16 (1024 + 8 pad -> bank rotate)
#define STILES 4      // s-tiles (of 16 rows) per workgroup in scores kernel

#define WMMA_BF16(A, Bm, C) \
  __builtin_amdgcn_wmma_f32_16x16x32_bf16(false, (A), false, (Bm), (short)0, (C), false, false)

// cheap accurate tanh: one v_exp_f32 + rcp; clamp avoids inf/inf
__device__ __forceinline__ float fast_tanh(float x) {
  float xc = fminf(fmaxf(x, -15.f), 15.f);
  float e  = __expf(2.f * xc);
  return (e - 1.f) / (e + 1.f);
}

// ---- f32 -> bf16 conversion (W1, W2 staged once into workspace) -----------
__global__ void __launch_bounds__(256) cvt_f32_bf16(
    const float* __restrict__ src, __bf16* __restrict__ dst, int n) {
  int i = blockIdx.x * 256 + threadIdx.x;
  if (i < n) dst[i] = (__bf16)src[i];
}

// ---------------------------------------------------------------------------
// 16-row WMMA core used by qproj (cost is <1% of total; simple version).
// Fragment layouts used here (16-bit, wave32, contiguous k for A and B):
//   A: lane L(<16)=row L holds K k0..k0+15; lane L+16 holds
//      k0+16..k0+31 -> two 16B ds_loads.
//   B: lane n(<16)=col n holds K k0..k0+15 -> contiguous 32B of a W row.
// ---------------------------------------------------------------------------
template<int KTOT>
__device__ __forceinline__ void wmma_block16(
    const float* __restrict__ Aglob, int arow_stride,
    const __bf16* __restrict__ Bmat, __bf16* As, v8f acc[4])
{
  const int tid   = threadIdx.x;
  const int lane  = tid & 31;
  const int wave  = tid >> 5;
  const int ucol  = wave * 64 + (lane & 15);
  const int arow  = lane & 15;
  const int kofsA = (lane >> 4) * 16;
  const int kofsB = (lane >> 4) * 16;

  for (int kb = 0; kb < KTOT; kb += 1024) {
#pragma unroll
    for (int i = 0; i < 16; ++i) {
      int idx4 = i * 256 + tid;
      int row  = idx4 >> 8;
      int c4   = idx4 & 255;
      float4 f = *(const float4*)(Aglob + (size_t)row * arow_stride + kb + c4 * 4);
      __bf16* d = &As[row * AROW + c4 * 4];
      d[0] = (__bf16)f.x; d[1] = (__bf16)f.y;
      d[2] = (__bf16)f.z; d[3] = (__bf16)f.w;
    }
    __syncthreads();
    for (int k0 = 0; k0 < 1024; k0 += 32) {
      v8bf alo = *(const v8bf*)&As[arow * AROW + k0 + kofsA];
      v8bf ahi = *(const v8bf*)&As[arow * AROW + k0 + kofsA + 8];
      v16bf a  = __builtin_shufflevector(alo, ahi,
                   0,1,2,3,4,5,6,7,8,9,10,11,12,13,14,15);
      const __bf16* bp = Bmat + (size_t)ucol * KTOT + kb + k0 + kofsB;
      v16bf b0 = *(const v16bf*)(bp);
      v16bf b1 = *(const v16bf*)(bp + 16 * KTOT);
      v16bf b2 = *(const v16bf*)(bp + 32 * KTOT);
      v16bf b3 = *(const v16bf*)(bp + 48 * KTOT);
      acc[0] = WMMA_BF16(a, b0, acc[0]);
      acc[1] = WMMA_BF16(a, b1, acc[1]);
      acc[2] = WMMA_BF16(a, b2, acc[2]);
      acc[3] = WMMA_BF16(a, b3, acc[3]);
    }
    __syncthreads();
  }
}

// ---- Kernel 1: q_proj[b,u] = q(viewed [256][2048]) @ W1^T + W1_b ----------
__global__ void __launch_bounds__(256) qproj_kernel(
    const float* __restrict__ dec, const __bf16* __restrict__ W1bf,
    const float* __restrict__ W1b, float* __restrict__ qp)
{
  __shared__ __bf16 As[16 * AROW];
  const int m0 = blockIdx.x * 16;
  v8f acc[4] = {};
  wmma_block16<KQ>(dec + (size_t)m0 * KQ, KQ, W1bf, As, acc);

  const int lane = threadIdx.x & 31;
  const int wave = threadIdx.x >> 5;
  const int un   = wave * 64 + (lane & 15);
  const int mrow = m0 + ((lane >> 4) << 3);   // C/D layout: VGPR r -> M=r / r+8
#pragma unroll
  for (int j = 0; j < 4; ++j) {
    int u = un + j * 16;
    float bias = W1b[u];
#pragma unroll
    for (int r = 0; r < 8; ++r)
      qp[(size_t)(mrow + r) * NU + u] = acc[j][r] + bias;
  }
}

// ---- Kernel 2: fused scores: Vw . tanh(qp + enc@W2^T + W2b) + Vb ----------
// WG = (b, 64 s-rows); wave owns 64 u-cols; B frags register-reused across
// the 4 s-tiles -> 4x less W2 traffic.
__global__ void __launch_bounds__(256) scores_kernel(
    const float* __restrict__ enc, const __bf16* __restrict__ W2bf,
    const float* __restrict__ qp,  const float* __restrict__ W2b,
    const float* __restrict__ Vw,  const float* __restrict__ Vb,
    float* __restrict__ scores)
{
  __shared__ __bf16 As[16 * STILES * AROW];   // 64 x 1032 bf16 ~ 129KB
  __shared__ float red[8][16 * STILES];
  const int s0 = blockIdx.x * (16 * STILES);
  const int b  = blockIdx.y;
  const float* Aglob = enc + ((size_t)b * NS + s0) * KE;

  const int tid   = threadIdx.x;
  const int lane  = tid & 31;
  const int wave  = tid >> 5;
  const int ucol  = wave * 64 + (lane & 15);
  const int arow  = lane & 15;
  const int kofsA = (lane >> 4) * 16;
  const int kofsB = (lane >> 4) * 16;

  v8f acc[STILES][4] = {};

  // Stage 64x1024 f32 tile -> bf16 LDS (coalesced float4 loads).
#pragma unroll 4
  for (int i = 0; i < 16 * STILES; ++i) {
    int idx4 = i * 256 + tid;
    int row  = idx4 >> 8;
    int c4   = idx4 & 255;
    float4 f = *(const float4*)(Aglob + (size_t)row * KE + c4 * 4);
    __bf16* d = &As[row * AROW + c4 * 4];
    d[0] = (__bf16)f.x; d[1] = (__bf16)f.y;
    d[2] = (__bf16)f.z; d[3] = (__bf16)f.w;
  }
  __syncthreads();

  for (int k0 = 0; k0 < KE; k0 += 32) {
    const __bf16* bp = W2bf + (size_t)ucol * KE + k0 + kofsB;
    v16bf b0 = *(const v16bf*)(bp);
    v16bf b1 = *(const v16bf*)(bp + 16 * KE);
    v16bf b2 = *(const v16bf*)(bp + 32 * KE);
    v16bf b3 = *(const v16bf*)(bp + 48 * KE);
#pragma unroll
    for (int st = 0; st < STILES; ++st) {
      const __bf16* ap = &As[(st * 16 + arow) * AROW + k0 + kofsA];
      v8bf alo = *(const v8bf*)ap;
      v8bf ahi = *(const v8bf*)(ap + 8);
      v16bf a  = __builtin_shufflevector(alo, ahi,
                   0,1,2,3,4,5,6,7,8,9,10,11,12,13,14,15);
      acc[st][0] = WMMA_BF16(a, b0, acc[st][0]);
      acc[st][1] = WMMA_BF16(a, b1, acc[st][1]);
      acc[st][2] = WMMA_BF16(a, b2, acc[st][2]);
      acc[st][3] = WMMA_BF16(a, b3, acc[st][3]);
    }
  }

  // Fused epilogue: tanh + dot with Vw, reduced over u.
  float part[STILES][8];
#pragma unroll
  for (int st = 0; st < STILES; ++st)
#pragma unroll
    for (int r = 0; r < 8; ++r) part[st][r] = 0.f;

#pragma unroll
  for (int j = 0; j < 4; ++j) {
    int u = ucol + j * 16;
    float qb = qp[b * NU + u] + W2b[u];
    float vw = Vw[u];
#pragma unroll
    for (int st = 0; st < STILES; ++st)
#pragma unroll
      for (int r = 0; r < 8; ++r)
        part[st][r] += vw * fast_tanh(acc[st][j][r] + qb);
  }
  // Deterministic reduction over the 16 u-lanes of each half-wave.
#pragma unroll
  for (int m = 1; m <= 8; m <<= 1)
#pragma unroll
    for (int st = 0; st < STILES; ++st)
#pragma unroll
      for (int r = 0; r < 8; ++r)
        part[st][r] += __shfl_xor(part[st][r], m, 32);

  if ((lane & 15) == 0) {
    int half = lane >> 4;
#pragma unroll
    for (int st = 0; st < STILES; ++st)
#pragma unroll
      for (int r = 0; r < 8; ++r)
        red[wave][st * 16 + half * 8 + r] = part[st][r];
  }
  __syncthreads();
  if (tid < 16 * STILES) {   // then across 8 waves, fixed order
    float s = Vb[0];
#pragma unroll
    for (int w = 0; w < 8; ++w) s += red[w][tid];
    scores[b * NS + s0 + tid] = s;
  }
}

// ---- Kernel 3: softmax over S per batch ----------------------------------
__global__ void __launch_bounds__(256) softmax_kernel(
    const float* __restrict__ sc, float* __restrict__ w)
{
  const int b = blockIdx.x, tid = threadIdx.x;
  const int lane = tid & 31, wave = tid >> 5;
  __shared__ float tmp[8];
  float v = sc[b * NS + tid];
  float m = v;
#pragma unroll
  for (int d = 16; d >= 1; d >>= 1) m = fmaxf(m, __shfl_xor(m, d, 32));
  if (lane == 0) tmp[wave] = m;
  __syncthreads();
  if (tid == 0) { float mm = tmp[0]; for (int i = 1; i < 8; ++i) mm = fmaxf(mm, tmp[i]); tmp[0] = mm; }
  __syncthreads();
  const float mm = tmp[0];
  __syncthreads();
  float e = expf(v - mm);
  float s = e;
#pragma unroll
  for (int d = 16; d >= 1; d >>= 1) s += __shfl_xor(s, d, 32);
  if (lane == 0) tmp[wave] = s;
  __syncthreads();
  if (tid == 0) { float t = 0.f; for (int i = 0; i < 8; ++i) t += tmp[i]; tmp[0] = t; }
  __syncthreads();
  w[b * NS + tid] = e / tmp[0];
}

// ---- Kernel 4: context[b,e] = sum_s w[b,s]*enc[b,s,e] (HBM-bound, f32) ---
__global__ void __launch_bounds__(256) context_kernel(
    const float* __restrict__ enc, const float* __restrict__ w,
    float* __restrict__ out)
{
  const int b = blockIdx.x, tid = threadIdx.x;
  __shared__ float wl[NS];
  wl[tid] = w[b * NS + tid];
  __syncthreads();
  const float4* ep = (const float4*)(enc + (size_t)b * NS * KE) + tid;
  float4 acc = make_float4(0.f, 0.f, 0.f, 0.f);
#pragma unroll 8
  for (int s = 0; s < NS; ++s) {
    float ws = wl[s];
    float4 v = ep[(size_t)s * 256];
    acc.x = fmaf(ws, v.x, acc.x);
    acc.y = fmaf(ws, v.y, acc.y);
    acc.z = fmaf(ws, v.z, acc.z);
    acc.w = fmaf(ws, v.w, acc.w);
  }
  *(float4*)(out + (size_t)b * KE + tid * 4) = acc;
}

// ---------------------------------------------------------------------------
extern "C" void kernel_launch(void* const* d_in, const int* in_sizes, int n_in,
                              void* d_out, int out_size, void* d_ws, size_t ws_size,
                              hipStream_t stream) {
  (void)in_sizes; (void)n_in; (void)out_size; (void)ws_size;
  const float* dec = (const float*)d_in[0];
  const float* enc = (const float*)d_in[1];
  const float* W1w = (const float*)d_in[2];
  const float* W1b = (const float*)d_in[3];
  const float* W2w = (const float*)d_in[4];
  const float* W2b = (const float*)d_in[5];
  const float* Vw  = (const float*)d_in[6];
  const float* Vb  = (const float*)d_in[7];
  float* out = (float*)d_out;

  char* ws = (char*)d_ws;                          // workspace layout (bytes):
  __bf16* W1bf = (__bf16*)(ws);                    // [512*2048] bf16 = 2 MB
  __bf16* W2bf = (__bf16*)(ws + 2097152);          // [512*1024] bf16 = 1 MB
  float*  qp   = (float*)(ws + 3145728);           // [256*512]  f32
  float*  sc   = (float*)(ws + 3670016);           // [256*256]  f32
  float*  wts  = out + NB * KE;                    // d_out: context | weights

  cvt_f32_bf16<<<4096, 256, 0, stream>>>(W1w, W1bf, NU * KQ);
  cvt_f32_bf16<<<2048, 256, 0, stream>>>(W2w, W2bf, NU * KE);
  qproj_kernel<<<16, 256, 0, stream>>>(dec, W1bf, W1b, qp);
  scores_kernel<<<dim3(NS / (16 * STILES), NB), 256, 0, stream>>>(
      enc, W2bf, qp, W2b, Vw, Vb, sc);
  softmax_kernel<<<NB, 256, 0, stream>>>(sc, wts);
  context_kernel<<<NB, 256, 0, stream>>>(enc, wts, out);
}
